// GNNLayer_5686536699929
// MI455X (gfx1250) — matrix operand order repaired and checked by reading, not compiled, not run on hardware
//
#include <hip/hip_runtime.h>

typedef __attribute__((ext_vector_type(16))) _Float16 v16h;
typedef __attribute__((ext_vector_type(8)))  _Float16 v8h;
typedef __attribute__((ext_vector_type(8)))  float    v8f;

__device__ __forceinline__ v8f wmma_f16(v16h a, v16h b, v8f c) {
  return __builtin_amdgcn_wmma_f32_16x16x32_f16(
      /*neg_a=*/false, a, /*neg_b=*/false, b,
      /*c_mod=*/(short)0, c, /*reuse_a=*/false, /*reuse_b=*/false);
}

__device__ __forceinline__ v16h mk16(v8h a, v8h b) {
  return __builtin_shufflevector(a, b, 0,1,2,3,4,5,6,7,8,9,10,11,12,13,14,15);
}

// B-matrix fragment loaded straight from the f32 weight matrix (row-major
// [out, in]); lane holds 16 consecutive K values of column n = row n of W.
__device__ __forceinline__ v16h ldB_f32(const float* __restrict__ W,
                                        int stride, int n, int k) {
  v8f lo = *(const v8f*)(W + n * stride + k);
  v8f hi = *(const v8f*)(W + n * stride + k + 8);
  return mk16(__builtin_convertvector(lo, v8h),
              __builtin_convertvector(hi, v8h));
}

__device__ __forceinline__ float silu(float v) {
  return v / (1.0f + __expf(-v));
}

// A-fragment chunk for the edge MLP: 8 consecutive f16 of the virtual row
// [ x[r] (0..63) | x[c] (64..127) | ef[e] (128..159) ].  kk is 8-aligned so a
// chunk never crosses a segment boundary.
__device__ __forceinline__ v8h ldA_edge(const _Float16* __restrict__ xh,
                                        const float* __restrict__ ef,
                                        int r, int c, int e, int kk) {
  if (kk < 64)  return *(const v8h*)(xh + r * 64 + kk);
  if (kk < 128) return *(const v8h*)(xh + c * 64 + (kk - 64));
  v8f f = *(const v8f*)(ef + e * 32 + (kk - 128));
  return __builtin_convertvector(f, v8h);
}

// A-fragment chunk for the node MLP: [ x[i] (0..63) | agg[i] (64..127) ]
__device__ __forceinline__ v8h ldA_node(const _Float16* __restrict__ xh,
                                        const float* __restrict__ agg,
                                        int i, int kk) {
  if (kk < 64) return *(const v8h*)(xh + i * 64 + kk);
  v8f f = *(const v8f*)(agg + i * 64 + (kk - 64));
  return __builtin_convertvector(f, v8h);
}

// ---------------------------------------------------------------------------
// prep: x(f32) -> xh(f16), agg = 0   (run every launch: deterministic)
// ---------------------------------------------------------------------------
__global__ void __launch_bounds__(256) prep_kernel(
    const float* __restrict__ x, _Float16* __restrict__ xh,
    float* __restrict__ agg, int n) {
  int i = blockIdx.x * 256 + threadIdx.x;
  if (i < n) {
    xh[i]  = (_Float16)x[i];
    agg[i] = 0.0f;
  }
}

// ---------------------------------------------------------------------------
// Edge MLP + scatter-add.  One wave per 16-edge tile (grid-stride over tiles).
// All weight fragments live in VGPRs; LDS used only for the per-wave h
// transpose (C-layout -> A-layout) and the scatter row indices.
// ---------------------------------------------------------------------------
__global__ void __launch_bounds__(256) edge_kernel(
    const int*   __restrict__ ei,     // [2, E]
    const float* __restrict__ ef,     // [E, 32]
    const float* __restrict__ We1,    // [64, 160]
    const float* __restrict__ be1,    // [64]
    const float* __restrict__ We2,    // [64, 64]
    const float* __restrict__ be2,    // [64]
    const _Float16* __restrict__ xh,  // [N, 64] f16
    float* __restrict__ agg,          // [N, 64] f32 (atomic accum)
    int nEdges) {
  __shared__ __align__(32) _Float16 sH[8][16 * 64];  // per-wave h staging
  __shared__ int sRow[8][16];

  const int tid   = threadIdx.x;
  const int lane  = tid & 31;
  const int wid   = tid >> 5;
  const int col   = lane & 15;       // B/C column, also A row (edge in tile)
  const int sel   = lane >> 4;       // 0: lanes 0-15, 1: lanes 16-31
  const int sel8  = sel * 8;
  const int sel16 = sel * 16;

  // ---- hoist all weight fragments + biases into registers ----
  v16h B1[5][4];                     // layer1: [kstep][ntile]
  v16h B2[2][4];                     // layer2
  float b1[4], b2[4];
#pragma unroll
  for (int s = 0; s < 5; ++s)
#pragma unroll
    for (int t = 0; t < 4; ++t)
      B1[s][t] = ldB_f32(We1, 160, t * 16 + col, s * 32 + sel16);
#pragma unroll
  for (int s = 0; s < 2; ++s)
#pragma unroll
    for (int t = 0; t < 4; ++t)
      B2[s][t] = ldB_f32(We2, 64, t * 16 + col, s * 32 + sel16);
#pragma unroll
  for (int t = 0; t < 4; ++t) {
    b1[t] = be1[t * 16 + col];
    b2[t] = be2[t * 16 + col];
  }

  const int nTiles    = (nEdges + 15) >> 4;
  const int waveId    = blockIdx.x * 8 + wid;
  const int waveCount = gridDim.x * 8;

  for (int tile = waveId; tile < nTiles; tile += waveCount) {
    const int base = tile * 16;

    // stage destination (source-node) indices for the scatter
    if (lane < 16) {
      int e = base + lane;
      sRow[wid][lane] = (e < nEdges) ? ei[e] : 0;
    }

    int eA = base + col;
    if (eA >= nEdges) eA = nEdges - 1;  // clamped; result discarded by guard
    const int rA = ei[eA];
    const int cA = ei[nEdges + eA];

    // prefetch next tile's streaming data
    int tnext = tile + waveCount;
    if (tnext < nTiles && lane == 0) {
      __builtin_prefetch(ei + tnext * 16, 0, 0);
      __builtin_prefetch(ef + (size_t)tnext * 16 * 32, 0, 0);
    }

    // -------- layer 1: [16 x 160] x [160 x 64] --------
    v8f acc[4] = {};
#pragma unroll
    for (int s = 0; s < 5; ++s) {
      const int k0 = s * 32;
      v8h c1 = ldA_edge(xh, ef, rA, cA, eA, k0 + sel16);
      v8h c2 = ldA_edge(xh, ef, rA, cA, eA, k0 + sel16 + 8);
      v16h a = mk16(c1, c2);
#pragma unroll
      for (int t = 0; t < 4; ++t)
        acc[t] = wmma_f16(a, B1[s][t], acc[t]);
    }

    // bias + SiLU, stage h (f16) via LDS to transpose C-layout -> A-layout
#pragma unroll
    for (int t = 0; t < 4; ++t) {
#pragma unroll
      for (int j = 0; j < 8; ++j) {
        float v = silu(acc[t][j] + b1[t]);
        sH[wid][(j + sel8) * 64 + (t * 16 + col)] = (_Float16)v;
      }
    }
    asm volatile("s_wait_dscnt 0" ::: "memory");

    // -------- layer 2: [16 x 64] x [64 x 64] --------
    v8f acc2[4] = {};
#pragma unroll
    for (int s = 0; s < 2; ++s) {
      const int k0 = s * 32;
      const _Float16* hp = &sH[wid][col * 64];
      v8h c1 = *(const v8h*)(hp + k0 + sel16);
      v8h c2 = *(const v8h*)(hp + k0 + sel16 + 8);
      v16h a = mk16(c1, c2);
#pragma unroll
      for (int t = 0; t < 4; ++t)
        acc2[t] = wmma_f16(a, B2[s][t], acc2[t]);
    }

    // bias + SiLU + scatter-add (one f32 atomic per output element)
#pragma unroll
    for (int t = 0; t < 4; ++t) {
      const int n = t * 16 + col;
#pragma unroll
      for (int j = 0; j < 8; ++j) {
        const int el = j + sel8;
        if (base + el < nEdges) {
          float m = silu(acc2[t][j] + b2[t]);
          __hip_atomic_fetch_add(&agg[sRow[wid][el] * 64 + n], m,
                                 __ATOMIC_RELAXED, __HIP_MEMORY_SCOPE_AGENT);
        }
      }
    }
  }
}

// ---------------------------------------------------------------------------
// Node MLP.  One wave per 16-node tile; weights register-resident.
// ---------------------------------------------------------------------------
__global__ void __launch_bounds__(256) node_kernel(
    const float* __restrict__ Wn1,    // [64, 128]
    const float* __restrict__ bn1,    // [64]
    const float* __restrict__ Wn2,    // [64, 64]
    const float* __restrict__ bn2,    // [64]
    const _Float16* __restrict__ xh,  // [N, 64] f16
    const float* __restrict__ agg,    // [N, 64] f32
    float* __restrict__ out,          // [N, 64] f32
    int nNodes) {
  __shared__ __align__(32) _Float16 sH[8][16 * 64];

  const int tid   = threadIdx.x;
  const int lane  = tid & 31;
  const int wid   = tid >> 5;
  const int col   = lane & 15;
  const int sel   = lane >> 4;
  const int sel8  = sel * 8;
  const int sel16 = sel * 16;

  v16h B1[4][4];
  v16h B2[2][4];
  float b1[4], b2[4];
#pragma unroll
  for (int s = 0; s < 4; ++s)
#pragma unroll
    for (int t = 0; t < 4; ++t)
      B1[s][t] = ldB_f32(Wn1, 128, t * 16 + col, s * 32 + sel16);
#pragma unroll
  for (int s = 0; s < 2; ++s)
#pragma unroll
    for (int t = 0; t < 4; ++t)
      B2[s][t] = ldB_f32(Wn2, 64, t * 16 + col, s * 32 + sel16);
#pragma unroll
  for (int t = 0; t < 4; ++t) {
    b1[t] = bn1[t * 16 + col];
    b2[t] = bn2[t * 16 + col];
  }

  const int nTiles    = (nNodes + 15) >> 4;
  const int waveId    = blockIdx.x * 8 + wid;
  const int waveCount = gridDim.x * 8;

  for (int tile = waveId; tile < nTiles; tile += waveCount) {
    const int base = tile * 16;
    int iA = base + col;
    if (iA >= nNodes) iA = nNodes - 1;

    // -------- layer 1: [16 x 128] x [128 x 64] --------
    v8f acc[4] = {};
#pragma unroll
    for (int s = 0; s < 4; ++s) {
      const int k0 = s * 32;
      v8h c1 = ldA_node(xh, agg, iA, k0 + sel16);
      v8h c2 = ldA_node(xh, agg, iA, k0 + sel16 + 8);
      v16h a = mk16(c1, c2);
#pragma unroll
      for (int t = 0; t < 4; ++t)
        acc[t] = wmma_f16(a, B1[s][t], acc[t]);
    }

#pragma unroll
    for (int t = 0; t < 4; ++t) {
#pragma unroll
      for (int j = 0; j < 8; ++j) {
        float v = silu(acc[t][j] + b1[t]);
        sH[wid][(j + sel8) * 64 + (t * 16 + col)] = (_Float16)v;
      }
    }
    asm volatile("s_wait_dscnt 0" ::: "memory");

    // -------- layer 2: [16 x 64] x [64 x 64], no final activation --------
    v8f acc2[4] = {};
#pragma unroll
    for (int s = 0; s < 2; ++s) {
      const int k0 = s * 32;
      const _Float16* hp = &sH[wid][col * 64];
      v8h c1 = *(const v8h*)(hp + k0 + sel16);
      v8h c2 = *(const v8h*)(hp + k0 + sel16 + 8);
      v16h a = mk16(c1, c2);
#pragma unroll
      for (int t = 0; t < 4; ++t)
        acc2[t] = wmma_f16(a, B2[s][t], acc2[t]);
    }

#pragma unroll
    for (int t = 0; t < 4; ++t) {
      const int n = t * 16 + col;
#pragma unroll
      for (int j = 0; j < 8; ++j) {
        const int el = j + sel8;
        if (base + el < nNodes)
          out[(base + el) * 64 + n] = acc2[t][j] + b2[t];
      }
    }
  }
}

// ---------------------------------------------------------------------------
extern "C" void kernel_launch(void* const* d_in, const int* in_sizes, int n_in,
                              void* d_out, int out_size, void* d_ws, size_t ws_size,
                              hipStream_t stream) {
  const float* x   = (const float*)d_in[0];
  const int*   ei  = (const int*)  d_in[1];
  const float* ef  = (const float*)d_in[2];
  const float* We1 = (const float*)d_in[3];
  const float* be1 = (const float*)d_in[4];
  const float* We2 = (const float*)d_in[5];
  const float* be2 = (const float*)d_in[6];
  const float* Wn1 = (const float*)d_in[7];
  const float* bn1 = (const float*)d_in[8];
  const float* Wn2 = (const float*)d_in[9];
  const float* bn2 = (const float*)d_in[10];

  const int nNodes = in_sizes[0] / 64;   // x: [N, 64]
  const int nEdges = in_sizes[2] / 32;   // edge_feat: [E, 32]

  // workspace layout: xh (f16, N*64) | agg (f32, N*64)
  _Float16* xh = (_Float16*)d_ws;
  float* agg = (float*)((char*)d_ws + (size_t)nNodes * 64 * sizeof(_Float16));

  const int nElem = nNodes * 64;
  prep_kernel<<<(nElem + 255) / 256, 256, 0, stream>>>(x, xh, agg, nElem);

  edge_kernel<<<512, 256, 0, stream>>>(ei, ef, We1, be1, We2, be2,
                                       xh, agg, nEdges);

  node_kernel<<<256, 256, 0, stream>>>(Wn1, bn1, Wn2, bn2,
                                       xh, agg, (float*)d_out, nNodes);
}
